// SparseTQDet_7284264534485
// MI455X (gfx1250) — matrix operand order repaired and checked by reading, not compiled, not run on hardware
//
#include <hip/hip_runtime.h>
#include <math.h>

typedef __bf16 v16bf __attribute__((ext_vector_type(16)));
typedef __bf16 v8bf  __attribute__((ext_vector_type(8)));
typedef float  v8f   __attribute__((ext_vector_type(8)));

#define NTOT 33600
#define IMGSZ 640.0f

static __device__ __forceinline__ v8f wmma_bf16(v16bf a, v16bf b, v8f c) {
  return __builtin_amdgcn_wmma_f32_16x16x32_bf16(false, a, false, b, (short)0, c, false, false);
}

// ---------------------------------------------------------------------------
// Weight packing: conv weights f32 [Co][128][3][3] -> bf16 A-fragments
// fragment layout [co_tile][kc(36)][lane(32)][e(16)], K-order = tap*128 + ci
// A 16x32 per-lane map used here (contiguous k; B reads match): lane l: m=l&15, half=l>>4;
//   K = half*16 + e
// ---------------------------------------------------------------------------
__global__ void pack_stem_w(const float* __restrict__ w, __bf16* __restrict__ out) {
  int i = blockIdx.x * 256 + threadIdx.x;
  if (i >= 8 * 36 * 32 * 16) return;
  int e = i & 15, lane = (i >> 4) & 31, kc = (i >> 9) % 36, ct = (i >> 9) / 36;
  int m = lane & 15, half = lane >> 4;
  int K = half * 16 + e;
  int t = kc >> 2, ci = ((kc & 3) << 5) + K;
  int co = ct * 16 + m;
  out[i] = (__bf16)w[(co * 128 + ci) * 9 + t];
}

__global__ void pack_head_w(const float* __restrict__ obj_w, const float* __restrict__ box_w,
                            __bf16* __restrict__ out) {
  int i = blockIdx.x * 256 + threadIdx.x;
  if (i >= 36 * 32 * 16) return;
  int e = i & 15, lane = (i >> 4) & 31, kc = (i >> 9);
  int m = lane & 15, half = lane >> 4;
  int K = half * 16 + e;
  int t = kc >> 2, ci = ((kc & 3) << 5) + K;
  float v = 0.f;
  if (m == 0)       v = obj_w[ci * 9 + t];
  else if (m <= 4)  v = box_w[((m - 1) * 128 + ci) * 9 + t];
  out[i] = (__bf16)v;
}

__global__ void zero_f(float* p, int n) {
  int i = blockIdx.x * 256 + threadIdx.x;
  if (i < n) p[i] = 0.f;
}

// ---------------------------------------------------------------------------
// 3x3 conv (SAME) as implicit GEMM via WMMA bf16. One block = 8 waves =
// all 128 output channels x 16 pixels of one row. Fused GN partial stats.
// ---------------------------------------------------------------------------
__global__ __launch_bounds__(256) void conv3x3_wmma(
    const float* __restrict__ in, const __bf16* __restrict__ wpack,
    float* __restrict__ out, float* __restrict__ gstats, int H, int W) {
  const int x0 = blockIdx.x * 16, y = blockIdx.y, b = blockIdx.z;
  const int tid = threadIdx.x, wave = tid >> 5, lane = tid & 31;
  __shared__ __bf16 tile[3 * 18 * 128];

  for (int p = tid; p < 384; p += 256) {         // (ci, row) pairs
    int ci = p & 127, r = p >> 7;
    int gy = y + r - 1;
    bool rowOK = (unsigned)gy < (unsigned)H;
    const float* rowp = in + ((b * 128 + ci) * H + (rowOK ? gy : 0)) * W;
    __bf16* drow = tile + (r * 18) * 128 + ci;
    for (int c = 0; c < 18; ++c) {
      int gx = x0 - 1 + c;
      float v = (rowOK && (unsigned)gx < (unsigned)W) ? rowp[gx] : 0.f;
      drow[c * 128] = (__bf16)v;
    }
  }
  __syncthreads();

  const int nN = lane & 15, kh = lane >> 4;
  v8f acc = {0.f, 0.f, 0.f, 0.f, 0.f, 0.f, 0.f, 0.f};
  for (int kc = 0; kc < 36; ++kc) {
    int t = kc >> 2, cib = (kc & 3) << 5;
    int dy = t / 3, dx = t % 3;
    v16bf a = *(const v16bf*)(wpack + (((wave * 36 + kc) * 32 + lane) << 4));
    int lc = nN + dx;
    v16bf bf = *(const v16bf*)(tile + (dy * 18 + lc) * 128 + cib + kh * 16);
    acc = wmma_bf16(a, bf, acc);
  }

  float s = 0.f, s2 = 0.f;
  if (x0 + nN < W) {
    #pragma unroll
    for (int i = 0; i < 8; ++i) {
      int co = wave * 16 + (kh << 3) + i;
      float v = acc[i];
      out[((b * 128 + co) * H + y) * W + x0 + nN] = v;
      s += v; s2 += v * v;
    }
  }
  for (int off = 16; off > 0; off >>= 1) {
    s  += __shfl_xor(s,  off, 32);
    s2 += __shfl_xor(s2, off, 32);
  }
  if (lane == 0) {
    atomicAdd(&gstats[(b * 8 + wave) * 2 + 0], s);
    atomicAdd(&gstats[(b * 8 + wave) * 2 + 1], s2);
  }
}

__global__ void gn_relu(float* __restrict__ buf, const float* __restrict__ stats,
                        const float* __restrict__ g, const float* __restrict__ be,
                        int H, int W) {
  long i = (long)blockIdx.x * 256 + threadIdx.x;
  int HW = H * W;
  long t = i / HW;
  int c = (int)(t & 127), b = (int)(t >> 7);
  int gi = c >> 4;
  float sum = stats[(b * 8 + gi) * 2], sq = stats[(b * 8 + gi) * 2 + 1];
  float cnt = 16.f * (float)HW;
  float mu = sum / cnt;
  float var = sq / cnt - mu * mu;
  float rs = rsqrtf(var + 1e-5f);
  float v = buf[i];
  v = (v - mu) * rs * g[c] + be[c];
  buf[i] = fmaxf(v, 0.f);
}

// ---------------------------------------------------------------------------
// Head conv: 1 WMMA co-tile (obj row 0, box rows 1..4), 8 waves = 128 px/row
// ---------------------------------------------------------------------------
__global__ __launch_bounds__(256) void head_conv_wmma(
    const float* __restrict__ in, const __bf16* __restrict__ wpack,
    float* __restrict__ objb, float* __restrict__ ltrb,
    const float* __restrict__ scales, const float* __restrict__ obj_b,
    const float* __restrict__ box_b, int lvl, int H, int W, int poff) {
  const int xg0 = blockIdx.x * 128, y = blockIdx.y, b = blockIdx.z;
  const int tid = threadIdx.x, wave = tid >> 5, lane = tid & 31;
  extern __shared__ char hsm[];
  __bf16* tile = (__bf16*)hsm;                       // [3][130][128]

  for (int p = tid; p < 384; p += 256) {
    int ci = p & 127, r = p >> 7;
    int gy = y + r - 1;
    bool rowOK = (unsigned)gy < (unsigned)H;
    const float* rowp = in + ((b * 128 + ci) * H + (rowOK ? gy : 0)) * W;
    __bf16* drow = tile + (r * 130) * 128 + ci;
    for (int c = 0; c < 130; ++c) {
      int gx = xg0 - 1 + c;
      float v = (rowOK && (unsigned)gx < (unsigned)W) ? rowp[gx] : 0.f;
      drow[c * 128] = (__bf16)v;
    }
  }
  __syncthreads();

  const int x0 = xg0 + wave * 16;
  if (x0 >= W) return;
  const int nN = lane & 15, kh = lane >> 4;
  v8f acc = {0.f, 0.f, 0.f, 0.f, 0.f, 0.f, 0.f, 0.f};
  for (int kc = 0; kc < 36; ++kc) {
    int t = kc >> 2, cib = (kc & 3) << 5;
    int dy = t / 3, dx = t % 3;
    v16bf a = *(const v16bf*)(wpack + ((kc * 32 + lane) << 4));
    int lc = wave * 16 + nN + dx;
    v16bf bf = *(const v16bf*)(tile + (dy * 130 + lc) * 128 + cib + kh * 16);
    acc = wmma_bf16(a, bf, acc);
  }
  int x = x0 + nN;
  if (kh == 0 && x < W) {
    float sc = scales[lvl];
    long p = (long)b * NTOT + poff + y * W + x;
    objb[p] = acc[0] + obj_b[0];
    #pragma unroll
    for (int i = 1; i <= 4; ++i)
      ltrb[p * 4 + (i - 1)] = fmaxf(sc * (acc[i] + box_b[i - 1]), 0.f);
  }
}

// ---------------------------------------------------------------------------
// Adaptive pool to 8x8 per level + mem positional encoding -> m_attn tokens
// ---------------------------------------------------------------------------
__global__ void pool_mem(const float* __restrict__ h2, const float* __restrict__ mpos_w,
                         const float* __restrict__ mpos_b, float* __restrict__ mem) {
  int id = blockIdx.x * 256 + threadIdx.x;            // 8*192*128
  int c = id & 127, t = (id >> 7) % 192, b = id / (192 * 128);
  int lvl = t >> 6, cell = t & 63, py = cell >> 3, px = cell & 7;
  int H, W; long off;
  if (lvl == 0)      { H = 160; W = 160; off = 0; }
  else if (lvl == 1) { H = 80;  W = 80;  off = 8L * 128 * 25600; }
  else               { H = 40;  W = 40;  off = 8L * 128 * 25600 + 8L * 128 * 6400; }
  int ch = H >> 3, cw = W >> 3;
  const float* p = h2 + off + ((long)(b * 128 + c) * H + py * ch) * W + px * cw;
  float sm = 0.f;
  for (int yy = 0; yy < ch; ++yy)
    for (int xx = 0; xx < cw; ++xx) sm += p[yy * W + xx];
  float pooled = sm / (float)(ch * cw);
  float cx = (px + 0.5f) * 0.125f, cy = (py + 0.5f) * 0.125f, sz = 0.125f;
  float pos = mpos_b[c] + cx * mpos_w[c * 4] + cy * mpos_w[c * 4 + 1]
            + sz * mpos_w[c * 4 + 2] + sz * mpos_w[c * 4 + 3];
  mem[((long)b * 192 + t) * 128 + c] = pooled + pos;
}

// ---------------------------------------------------------------------------
// Top-K(64) per batch: min/max, 4096-bin histogram, threshold collect,
// wave0 selection sort with tie-break on smaller index.
// ---------------------------------------------------------------------------
__global__ __launch_bounds__(1024) void topk_kernel(float* __restrict__ scores,
                                                    int* __restrict__ topk_idx) {
  const int b = blockIdx.x, tid = threadIdx.x;
  float* s = scores + (long)b * NTOT;
  __shared__ float red[1024];
  __shared__ int hist[4096];
  __shared__ float cval[1024];
  __shared__ int cidx[1024];
  __shared__ int ctr[2];
  __shared__ float mnmx[2];

  float mx = -1e30f, mn = 1e30f;
  for (int i = tid; i < NTOT; i += 1024) { float v = s[i]; mx = fmaxf(mx, v); mn = fminf(mn, v); }
  red[tid] = mx; __syncthreads();
  for (int st = 512; st > 0; st >>= 1) { if (tid < st) red[tid] = fmaxf(red[tid], red[tid + st]); __syncthreads(); }
  if (tid == 0) mnmx[0] = red[0];
  __syncthreads();
  red[tid] = mn; __syncthreads();
  for (int st = 512; st > 0; st >>= 1) { if (tid < st) red[tid] = fminf(red[tid], red[tid + st]); __syncthreads(); }
  if (tid == 0) mnmx[1] = red[0];
  __syncthreads();
  float vmax = mnmx[0], vmin = mnmx[1];
  float rng = fmaxf(vmax - vmin, 1e-12f);

  for (int i = tid; i < 4096; i += 1024) hist[i] = 0;
  __syncthreads();
  for (int i = tid; i < NTOT; i += 1024) {
    int bk = (int)((s[i] - vmin) * (4096.f / rng));
    bk = min(max(bk, 0), 4095);
    atomicAdd(&hist[bk], 1);
  }
  __syncthreads();
  if (tid == 0) {
    int cum = 0, tb = 0;
    for (int i = 4095; i >= 0; --i) { cum += hist[i]; if (cum >= 64) { tb = i; break; } }
    ctr[0] = max(tb - 1, 0);    // one bin of margin against fp binning edges
    ctr[1] = 0;
  }
  __syncthreads();
  float thresh = vmin + (float)ctr[0] * (rng / 4096.f);
  for (int i = tid; i < NTOT; i += 1024) {
    float v = s[i];
    if (v >= thresh) { int p = atomicAdd(&ctr[1], 1); if (p < 1024) { cval[p] = v; cidx[p] = i; } }
  }
  __syncthreads();
  int cc = min(ctr[1], 1024);
  if (tid < 32) {
    for (int k = 0; k < 64; ++k) {
      float bv = -1e30f; int bi = 0x7fffffff;
      for (int i = tid; i < cc; i += 32) {
        float v = cval[i]; int id = cidx[i];
        if (v > bv || (v == bv && id < bi)) { bv = v; bi = id; }
      }
      for (int off = 16; off > 0; off >>= 1) {
        float ov = __shfl_xor(bv, off, 32); int oi = __shfl_xor(bi, off, 32);
        if (ov > bv || (ov == bv && oi < bi)) { bv = ov; bi = oi; }
      }
      if (tid == 0) topk_idx[b * 64 + k] = bi;
      for (int i = tid; i < cc; i += 32) if (cidx[i] == bi) cval[i] = -1e30f;
      __builtin_amdgcn_wave_barrier();
    }
  }
}

// ---------------------------------------------------------------------------
// Gather selected features, decode LTRB boxes
// ---------------------------------------------------------------------------
__global__ void gather_kernel(const int* __restrict__ topk_idx, const float* __restrict__ ltrb,
                              const float* __restrict__ h2, float* __restrict__ boxes,
                              float* __restrict__ qbuf) {
  const int b = blockIdx.x, k = threadIdx.x;
  int idx = topk_idx[b * 64 + k];
  int H, W, st; long off; int poff;
  if (idx < 25600)      { H = 160; W = 160; st = 4;  off = 0; poff = 0; }
  else if (idx < 32000) { H = 80;  W = 80;  st = 8;  off = 8L * 128 * 25600; poff = 25600; }
  else                  { H = 40;  W = 40;  st = 16; off = 8L * 128 * 25600 + 8L * 128 * 6400; poff = 32000; }
  int p = idx - poff;
  int y = p / W, x = p % W;
  float px = (x + 0.5f) * st, py = (y + 0.5f) * st;
  const float* lt = ltrb + ((long)b * NTOT + idx) * 4;
  float* bx = boxes + (b * 64 + k) * 4;
  bx[0] = fminf(fmaxf(px - lt[0], 0.f), IMGSZ);
  bx[1] = fminf(fmaxf(py - lt[1], 0.f), IMGSZ);
  bx[2] = fminf(fmaxf(px + lt[2], 0.f), IMGSZ);
  bx[3] = fminf(fmaxf(py + lt[3], 0.f), IMGSZ);
  float* q = qbuf + ((long)b * 64 + k) * 128;
  for (int c = 0; c < 128; ++c)
    q[c] = h2[off + ((long)(b * 128 + c) * H + y) * W + x];
}

// ---------------------------------------------------------------------------
// Generic block GEMM on LDS-staged bf16 operands: O = act(sA @ sW^T * scale + bias)
// ---------------------------------------------------------------------------
__device__ __forceinline__ void stage(__bf16* dst, int rowsP, int colsP,
                                      const float* src, int rows, int cols,
                                      int srs, int scs, int tid) {
  int tot = rowsP * colsP;
  for (int i = tid; i < tot; i += 256) {
    int r = i / colsP, c = i - r * colsP;
    float v = (r < rows && c < cols) ? src[r * srs + c * scs] : 0.f;
    dst[i] = (__bf16)v;
  }
}

__device__ __forceinline__ void block_gemm(const __bf16* sA, const __bf16* sW,
                                           float* O, int ldo, int M, int N, int K,
                                           const float* bias, float scale, int act, int tid) {
  const int wave = tid >> 5, lane = tid & 31;
  const int nIn = lane & 15, kh = lane >> 4;
  const int nt = N >> 4, tiles = (M >> 4) * nt;
  for (int t = wave; t < tiles; t += 8) {
    const int m0 = (t / nt) << 4, n0 = (t - (t / nt) * nt) << 4;
    v8f acc = {0.f, 0.f, 0.f, 0.f, 0.f, 0.f, 0.f, 0.f};
    const __bf16* arow = sA + (m0 + nIn) * K;
    const __bf16* brow = sW + (n0 + nIn) * K;
    for (int kc = 0; kc < K; kc += 32) {
      v8bf alo = *(const v8bf*)(arow + kc + kh * 16);
      v8bf ahi = *(const v8bf*)(arow + kc + kh * 16 + 8);
      v16bf a;
      #pragma unroll
      for (int j = 0; j < 8; ++j) { a[j] = alo[j]; a[j + 8] = ahi[j]; }
      v16bf bf = *(const v16bf*)(brow + kc + kh * 16);
      acc = wmma_bf16(a, bf, acc);
    }
    const int n = n0 + nIn;
    const float bv = bias ? bias[n] : 0.f;
    #pragma unroll
    for (int i = 0; i < 8; ++i) {
      int m = m0 + (kh << 3) + i;
      float v = acc[i] * scale + bv;
      if (act == 1)      v = fmaxf(v, 0.f);
      else if (act == 2) v = 0.5f * v * (1.f + erff(v * 0.70710678f));
      O[m * ldo + n] = v;
    }
  }
}

struct TParams {
  const float *qpos_w, *qpos_b;
  const float *sa_in_w, *sa_in_b, *sa_out_w, *sa_out_b;
  const float *ca_in_w, *ca_in_b, *ca_out_w, *ca_out_b;
  const float *ln1_g, *ln1_b, *ln2_g, *ln2_b, *ln3_g, *ln3_b;
  const float *ffn_w1, *ffn_b1, *ffn_w2, *ffn_b2, *delta_w, *delta_b;
  const float *cls_w, *cls_b, *ctr_w, *ctr_b, *iou_w, *iou_b, *unc_w, *unc_b;
  float *boxes, *qbuf, *mem, *scratch, *out;
};

__device__ __forceinline__ void ln_rows(float* q, const float* add,
                                        const float* g, const float* be, int tid) {
  if (tid < 64) {
    float* qr = q + tid * 128;
    const float* ar = add + tid * 128;
    float s = 0.f;
    for (int j = 0; j < 128; ++j) s += qr[j] + ar[j];
    float mu = s * (1.f / 128.f);
    float v = 0.f;
    for (int j = 0; j < 128; ++j) { float d = qr[j] + ar[j] - mu; v += d * d; }
    float rs = rsqrtf(v * (1.f / 128.f) + 1e-5f);
    for (int j = 0; j < 128; ++j) qr[j] = (qr[j] + ar[j] - mu) * rs * g[j] + be[j];
  }
}

#define TSB 237824L

__global__ __launch_bounds__(256) void refine_kernel(TParams P) {
  const int b = blockIdx.x, tid = threadIdx.x;
  extern __shared__ char smem[];
  __bf16* sA = (__bf16*)smem;                 // 32768 halves
  __bf16* sW = (__bf16*)(smem + 65536);       // 32768 halves
  float* q = P.qbuf + (long)b * 64 * 128;
  float* boxes = P.boxes + b * 64 * 4;
  const float* mem = P.mem + (long)b * 192 * 128;
  float* SC = P.scratch + (long)b * TSB;
  float* QP = SC;            float* QA = QP + 8192;   float* QKV = QA + 8192;
  float* MKV = QKV + 24576;  float* PROB = MKV + 49152;
  float* ATTO = PROB + 98304; float* TMP = ATTO + 8192;
  float* HID = TMP + 8192;   float* DLT = HID + 32768;

  for (int l = 0; l < 4; ++l) {
    // qpos & qa = q + qpos
    for (int i = tid; i < 8192; i += 256) {
      int m = i >> 7, c = i & 127;
      const float* bx = boxes + m * 4;
      float cx = 0.5f * (bx[0] + bx[2]) * (1.f / IMGSZ);
      float cy = 0.5f * (bx[1] + bx[3]) * (1.f / IMGSZ);
      float bw = fmaxf(bx[2] - bx[0], 0.f) * (1.f / IMGSZ);
      float bh = fmaxf(bx[3] - bx[1], 0.f) * (1.f / IMGSZ);
      float v = P.qpos_b[c] + cx * P.qpos_w[c * 4] + cy * P.qpos_w[c * 4 + 1]
              + bw * P.qpos_w[c * 4 + 2] + bh * P.qpos_w[c * 4 + 3];
      QP[i] = v; QA[i] = q[i] + v;
    }
    __syncthreads();

    // ---- self-attention in_proj (N split 384 -> 2x192) ----
    const float* w = P.sa_in_w + (long)l * 384 * 128;
    const float* bb = P.sa_in_b + l * 384;
    stage(sA, 64, 128, QA, 64, 128, 128, 1, tid);
    stage(sW, 192, 128, w, 192, 128, 128, 1, tid); __syncthreads();
    block_gemm(sA, sW, QKV, 384, 64, 192, 128, bb, 1.f, 0, tid); __syncthreads();
    stage(sW, 192, 128, w + 192 * 128, 192, 128, 128, 1, tid); __syncthreads();
    block_gemm(sA, sW, QKV + 192, 384, 64, 192, 128, bb + 192, 1.f, 0, tid); __syncthreads();

    // ---- SA scores per head (K=16 zero-padded to 32) ----
    for (int h = 0; h < 8; ++h) {
      stage(sA, 64, 32, QKV + h * 16, 64, 16, 384, 1, tid);
      stage(sW, 64, 32, QKV + 128 + h * 16, 64, 16, 384, 1, tid);
      __syncthreads();
      block_gemm(sA, sW, PROB + h * 64 * 64, 64, 64, 64, 32, nullptr, 0.25f, 0, tid);
      __syncthreads();
    }
    for (int r = tid; r < 512; r += 256) {
      float* row = PROB + r * 64;
      float mx = -1e30f;
      for (int j = 0; j < 64; ++j) mx = fmaxf(mx, row[j]);
      float sm = 0.f;
      for (int j = 0; j < 64; ++j) { float e = __expf(row[j] - mx); row[j] = e; sm += e; }
      float inv = 1.f / sm;
      for (int j = 0; j < 64; ++j) row[j] *= inv;
    }
    __syncthreads();
    for (int h = 0; h < 8; ++h) {
      stage(sA, 64, 64, PROB + h * 64 * 64, 64, 64, 64, 1, tid);
      stage(sW, 16, 64, QKV + 256 + h * 16, 16, 64, 1, 384, tid);   // V^T
      __syncthreads();
      block_gemm(sA, sW, ATTO + h * 16, 128, 64, 16, 64, nullptr, 1.f, 0, tid);
      __syncthreads();
    }
    stage(sA, 64, 128, ATTO, 64, 128, 128, 1, tid);
    stage(sW, 128, 128, P.sa_out_w + (long)l * 128 * 128, 128, 128, 128, 1, tid);
    __syncthreads();
    block_gemm(sA, sW, TMP, 128, 64, 128, 128, P.sa_out_b + l * 128, 1.f, 0, tid);
    __syncthreads();
    ln_rows(q, TMP, P.ln1_g + l * 128, P.ln1_b + l * 128, tid);
    __syncthreads();

    // ---- cross-attention ----
    for (int i = tid; i < 8192; i += 256) QA[i] = q[i] + QP[i];
    __syncthreads();
    const float* cw = P.ca_in_w + (long)l * 384 * 128;
    const float* cb = P.ca_in_b + l * 384;
    stage(sA, 64, 128, QA, 64, 128, 128, 1, tid);
    stage(sW, 128, 128, cw, 128, 128, 128, 1, tid); __syncthreads();
    block_gemm(sA, sW, QKV, 384, 64, 128, 128, cb, 1.f, 0, tid); __syncthreads();
    stage(sA, 192, 128, mem, 192, 128, 128, 1, tid);
    stage(sW, 256, 128, cw + 128 * 128, 256, 128, 128, 1, tid); __syncthreads();
    block_gemm(sA, sW, MKV, 256, 192, 256, 128, cb + 128, 1.f, 0, tid); __syncthreads();

    for (int h = 0; h < 8; ++h) {
      stage(sA, 64, 32, QKV + h * 16, 64, 16, 384, 1, tid);
      stage(sW, 192, 32, MKV + h * 16, 192, 16, 256, 1, tid);
      __syncthreads();
      block_gemm(sA, sW, PROB + h * 64 * 192, 192, 64, 192, 32, nullptr, 0.25f, 0, tid);
      __syncthreads();
    }
    for (int r = tid; r < 512; r += 256) {
      float* row = PROB + r * 192;
      float mx = -1e30f;
      for (int j = 0; j < 192; ++j) mx = fmaxf(mx, row[j]);
      float sm = 0.f;
      for (int j = 0; j < 192; ++j) { float e = __expf(row[j] - mx); row[j] = e; sm += e; }
      float inv = 1.f / sm;
      for (int j = 0; j < 192; ++j) row[j] *= inv;
    }
    __syncthreads();
    for (int h = 0; h < 8; ++h) {
      stage(sA, 64, 192, PROB + h * 64 * 192, 64, 192, 192, 1, tid);
      stage(sW, 16, 192, MKV + 128 + h * 16, 16, 192, 1, 256, tid);
      __syncthreads();
      block_gemm(sA, sW, ATTO + h * 16, 128, 64, 16, 192, nullptr, 1.f, 0, tid);
      __syncthreads();
    }
    stage(sA, 64, 128, ATTO, 64, 128, 128, 1, tid);
    stage(sW, 128, 128, P.ca_out_w + (long)l * 128 * 128, 128, 128, 128, 1, tid);
    __syncthreads();
    block_gemm(sA, sW, TMP, 128, 64, 128, 128, P.ca_out_b + l * 128, 1.f, 0, tid);
    __syncthreads();
    ln_rows(q, TMP, P.ln2_g + l * 128, P.ln2_b + l * 128, tid);
    __syncthreads();

    // ---- FFN ----
    const float* f1 = P.ffn_w1 + (long)l * 512 * 128;
    stage(sA, 64, 128, q, 64, 128, 128, 1, tid);
    stage(sW, 256, 128, f1, 256, 128, 128, 1, tid); __syncthreads();
    block_gemm(sA, sW, HID, 512, 64, 256, 128, P.ffn_b1 + l * 512, 1.f, 2, tid); __syncthreads();
    stage(sW, 256, 128, f1 + 256 * 128, 256, 128, 128, 1, tid); __syncthreads();
    block_gemm(sA, sW, HID + 256, 512, 64, 256, 128, P.ffn_b1 + l * 512 + 256, 1.f, 2, tid); __syncthreads();
    const float* f2 = P.ffn_w2 + (long)l * 128 * 512;
    stage(sA, 64, 512, HID, 64, 512, 512, 1, tid);
    stage(sW, 64, 512, f2, 64, 512, 512, 1, tid); __syncthreads();
    block_gemm(sA, sW, TMP, 128, 64, 64, 512, P.ffn_b2 + l * 128, 1.f, 0, tid); __syncthreads();
    stage(sW, 64, 512, f2 + 64 * 512, 64, 512, 512, 1, tid); __syncthreads();
    block_gemm(sA, sW, TMP + 64, 128, 64, 64, 512, P.ffn_b2 + l * 128 + 64, 1.f, 0, tid); __syncthreads();
    ln_rows(q, TMP, P.ln3_g + l * 128, P.ln3_b + l * 128, tid);
    __syncthreads();

    // ---- box deltas ----
    {
      int m = tid >> 2, n = tid & 3;
      const float* dw = P.delta_w + ((long)l * 4 + n) * 128;
      float s = P.delta_b[l * 4 + n];
      for (int k2 = 0; k2 < 128; ++k2) s += q[m * 128 + k2] * dw[k2];
      DLT[m * 4 + n] = s;
    }
    __syncthreads();
    if (tid < 64) {
      float* bx = boxes + tid * 4;
      float x1 = bx[0], y1 = bx[1], x2 = bx[2], y2 = bx[3];
      float wd = fmaxf(x2 - x1, 1.f), ht = fmaxf(y2 - y1, 1.f);
      float cx = 0.5f * (x1 + x2), cy = 0.5f * (y1 + y2);
      const float* d = DLT + tid * 4;
      float dx = tanhf(d[0]) * 0.5f, dy = tanhf(d[1]) * 0.5f;
      float dw2 = tanhf(d[2]) * 0.4f, dh2 = tanhf(d[3]) * 0.4f;
      float cx2 = cx + dx * wd, cy2 = cy + dy * ht;
      float w2 = wd * __expf(dw2), h2v = ht * __expf(dh2);
      bx[0] = fminf(fmaxf(cx2 - 0.5f * w2, 0.f), IMGSZ);
      bx[1] = fminf(fmaxf(cy2 - 0.5f * h2v, 0.f), IMGSZ);
      bx[2] = fminf(fmaxf(cx2 + 0.5f * w2, 0.f), IMGSZ);
      bx[3] = fminf(fmaxf(cy2 + 0.5f * h2v, 0.f), IMGSZ);
    }
    __syncthreads();
  }

  // ---- final scoring heads ----
  if (tid < 64) {
    const float* qr = q + tid * 128;
    float a = P.cls_b[0], c2 = P.ctr_b[0], io = P.iou_b[0], un = P.unc_b[0];
    for (int k2 = 0; k2 < 128; ++k2) {
      float v = qr[k2];
      a += v * P.cls_w[k2]; c2 += v * P.ctr_w[k2];
      io += v * P.iou_w[k2]; un += v * P.unc_w[k2];
    }
    float sa = 1.f / (1.f + __expf(-a)), sc = 1.f / (1.f + __expf(-c2));
    float si = 1.f / (1.f + __expf(-io)), su = 1.f / (1.f + __expf(-un));
    float score = sa * sc * si * (1.f - su);
    float* o = P.out + ((long)b * 64 + tid) * 5;
    const float* bx = boxes + tid * 4;
    o[0] = bx[0]; o[1] = bx[1]; o[2] = bx[2]; o[3] = bx[3]; o[4] = score;
  }
}

// ---------------------------------------------------------------------------
extern "C" void kernel_launch(void* const* d_in, const int* in_sizes, int n_in,
                              void* d_out, int out_size, void* d_ws, size_t ws_size,
                              hipStream_t stream) {
  const float* f[3] = {(const float*)d_in[0], (const float*)d_in[1], (const float*)d_in[2]};
  const float* stem1_w = (const float*)d_in[3];
  const float* gn1_g = (const float*)d_in[4];  const float* gn1_b = (const float*)d_in[5];
  const float* stem2_w = (const float*)d_in[6];
  const float* gn2_g = (const float*)d_in[7];  const float* gn2_b = (const float*)d_in[8];
  const float* obj_w = (const float*)d_in[9];  const float* obj_b = (const float*)d_in[10];
  const float* box_w = (const float*)d_in[11]; const float* box_b = (const float*)d_in[12];
  const float* scales = (const float*)d_in[13];

  // workspace carve
  char* ws = (char*)d_ws;
  size_t o = 0;
  auto carveF = [&](size_t n) -> float* { float* p = (float*)(ws + o); o += n * sizeof(float); return p; };
  float* bufA  = carveF(34406400UL);
  float* bufB  = carveF(34406400UL);
  float* objb  = carveF(8UL * NTOT);
  float* ltrb  = carveF(8UL * NTOT * 4);
  float* memb  = carveF(8UL * 192 * 128);
  float* qb    = carveF(8UL * 64 * 128);
  float* boxes = carveF(2048);
  float* stats = carveF(768);
  float* tsc   = carveF(8UL * TSB);
  int*   tkidx = (int*)carveF(512);
  o = (o + 255) & ~(size_t)255;
  __bf16* wp1 = (__bf16*)(ws + o); o += 147456 * 2;
  __bf16* wp2 = (__bf16*)(ws + o); o += 147456 * 2;
  __bf16* wph = (__bf16*)(ws + o); o += 18432 * 2;

  const int LH[3] = {160, 80, 40}, LW[3] = {160, 80, 40};
  const long LOFF[3] = {0, 8L * 128 * 25600, 8L * 128 * 25600 + 8L * 128 * 6400};
  const int LPOFF[3] = {0, 25600, 32000};

  zero_f<<<3, 256, 0, stream>>>(stats, 768);
  pack_stem_w<<<576, 256, 0, stream>>>(stem1_w, wp1);
  pack_stem_w<<<576, 256, 0, stream>>>(stem2_w, wp2);
  pack_head_w<<<72, 256, 0, stream>>>(obj_w, box_w, wph);

  for (int l = 0; l < 3; ++l) {
    dim3 g((LW[l] + 15) / 16, LH[l], 8);
    conv3x3_wmma<<<g, 256, 0, stream>>>(f[l], wp1, bufA + LOFF[l], stats + l * 128, LH[l], LW[l]);
  }
  for (int l = 0; l < 3; ++l) {
    int blocks = (int)(8L * 128 * LH[l] * LW[l] / 256);
    gn_relu<<<blocks, 256, 0, stream>>>(bufA + LOFF[l], stats + l * 128, gn1_g, gn1_b, LH[l], LW[l]);
  }
  for (int l = 0; l < 3; ++l) {
    dim3 g((LW[l] + 15) / 16, LH[l], 8);
    conv3x3_wmma<<<g, 256, 0, stream>>>(bufA + LOFF[l], wp2, bufB + LOFF[l], stats + (3 + l) * 128, LH[l], LW[l]);
  }
  for (int l = 0; l < 3; ++l) {
    int blocks = (int)(8L * 128 * LH[l] * LW[l] / 256);
    gn_relu<<<blocks, 256, 0, stream>>>(bufB + LOFF[l], stats + (3 + l) * 128, gn2_g, gn2_b, LH[l], LW[l]);
  }
  for (int l = 0; l < 3; ++l) {
    dim3 g((LW[l] + 127) / 128, LH[l], 8);
    head_conv_wmma<<<g, 256, 3 * 130 * 128 * 2, stream>>>(bufB + LOFF[l], wph, objb, ltrb,
                                                          scales, obj_b, box_b, l, LH[l], LW[l], LPOFF[l]);
  }
  pool_mem<<<768, 256, 0, stream>>>(bufB, (const float*)d_in[16], (const float*)d_in[17], memb);
  topk_kernel<<<8, 1024, 0, stream>>>(objb, tkidx);
  gather_kernel<<<8, 64, 0, stream>>>(tkidx, ltrb, bufB, boxes, qb);

  TParams P;
  P.qpos_w = (const float*)d_in[14]; P.qpos_b = (const float*)d_in[15];
  P.sa_in_w = (const float*)d_in[18]; P.sa_in_b = (const float*)d_in[19];
  P.sa_out_w = (const float*)d_in[20]; P.sa_out_b = (const float*)d_in[21];
  P.ca_in_w = (const float*)d_in[22]; P.ca_in_b = (const float*)d_in[23];
  P.ca_out_w = (const float*)d_in[24]; P.ca_out_b = (const float*)d_in[25];
  P.ln1_g = (const float*)d_in[26]; P.ln1_b = (const float*)d_in[27];
  P.ln2_g = (const float*)d_in[28]; P.ln2_b = (const float*)d_in[29];
  P.ln3_g = (const float*)d_in[30]; P.ln3_b = (const float*)d_in[31];
  P.ffn_w1 = (const float*)d_in[32]; P.ffn_b1 = (const float*)d_in[33];
  P.ffn_w2 = (const float*)d_in[34]; P.ffn_b2 = (const float*)d_in[35];
  P.delta_w = (const float*)d_in[36]; P.delta_b = (const float*)d_in[37];
  P.cls_w = (const float*)d_in[38]; P.cls_b = (const float*)d_in[39];
  P.ctr_w = (const float*)d_in[40]; P.ctr_b = (const float*)d_in[41];
  P.iou_w = (const float*)d_in[42]; P.iou_b = (const float*)d_in[43];
  P.unc_w = (const float*)d_in[44]; P.unc_b = (const float*)d_in[45];
  P.boxes = boxes; P.qbuf = qb; P.mem = memb; P.scratch = tsc; P.out = (float*)d_out;
  refine_kernel<<<8, 256, 131072, stream>>>(P);
  (void)in_sizes; (void)n_in; (void)out_size; (void)ws_size;
}
